// Attention_74577812128347
// MI455X (gfx1250) — matrix operand order repaired and checked by reading, not compiled, not run on hardware
//
#include <hip/hip_runtime.h>
#include <hip/hip_bf16.h>
#include <math.h>

typedef __attribute__((ext_vector_type(16))) _Float16 v16h;
typedef __attribute__((ext_vector_type(8)))  float    v8f;
typedef __attribute__((ext_vector_type(4)))  unsigned int v4u;
typedef __attribute__((ext_vector_type(8)))  int      v8i;
typedef __attribute__((ext_vector_type(4)))  int      v4i;

#define NHEADS 16
#define DHEAD  64
#define DMODEL 1024
#define EMB    1024          /* NHEADS*DHEAD */
#define BATCH  2
#define SEQ    2048
#define BLROWS (BATCH*SEQ)   /* 4096 */
#define TAUF   30.0f

#if __has_builtin(__builtin_amdgcn_tensor_load_to_lds) && \
    __has_builtin(__builtin_amdgcn_s_wait_tensorcnt)
#define USE_TDM 1
#endif

// ---------------------------------------------------------------------------
// Stage 1: fused QKV projection.  C = X(4096x1024) * W^T(1024x1024), f16 WMMA
// with f32 accumulation; result stored as f16 for stage 2 operands.
// Block = 128 threads (4 waves), 128x64 tile; wave w -> rows w*32..+31
// (2 A-fragments, so each B fragment read feeds 2 WMMAs).
// ---------------------------------------------------------------------------
__global__ __launch_bounds__(128)
void qkv_gemm_kernel(const float* __restrict__ X,
                     const float* __restrict__ Wq,
                     const float* __restrict__ Wk,
                     const float* __restrict__ Wv,
                     _Float16* __restrict__ ws_qkv)
{
    __shared__ _Float16 lds_x[128][32];  // activation tile  [row][k]
    __shared__ _Float16 lds_w[64][32];   // weight tile      [out-col][k]

    const int tid  = threadIdx.x;
    const int wv   = tid >> 5;          // wave 0..3
    const int lane = tid & 31;
    const int hl   = lane >> 4;         // half-wave 0/1
    const int ln   = lane & 15;

    const int colbase = blockIdx.x * 64;
    const int rowbase = blockIdx.y * 128;
    const int which   = blockIdx.z;     // 0=Q 1=K 2=V

    const float* W = (which == 0) ? Wq : (which == 1) ? Wk : Wv;
    _Float16* Out  = ws_qkv + (size_t)which * BLROWS * EMB;

    v8f acc[2][4] = {};

    const int wrow = tid >> 1;          // weight staging row 0..63
    const int wcol = (tid & 1) * 16;    // weight staging k offset

    for (int k0 = 0; k0 < DMODEL; k0 += 32) {
        // ---- stage X (one full row of 32 per thread) and W tiles ----
        {
            const float* xs = X + (size_t)(rowbase + tid) * DMODEL + k0;
            #pragma unroll
            for (int i = 0; i < 32; i += 4) {
                float4 f = *(const float4*)(xs + i);
                lds_x[tid][i+0] = (_Float16)f.x;
                lds_x[tid][i+1] = (_Float16)f.y;
                lds_x[tid][i+2] = (_Float16)f.z;
                lds_x[tid][i+3] = (_Float16)f.w;
            }
            const float* wsp = W + (size_t)(colbase + wrow) * DMODEL + k0 + wcol;
            #pragma unroll
            for (int i = 0; i < 16; i += 4) {
                float4 f = *(const float4*)(wsp + i);
                lds_w[wrow][wcol+i+0] = (_Float16)f.x;
                lds_w[wrow][wcol+i+1] = (_Float16)f.y;
                lds_w[wrow][wcol+i+2] = (_Float16)f.z;
                lds_w[wrow][wcol+i+3] = (_Float16)f.w;
            }
        }
        __syncthreads();

        // ---- 2 A fragments (16x32 each): lanes<16 K=0..15,
        //      lanes>=16 K=16..31; rows = ln on both halves ----
        v16h a[2];
        #pragma unroll
        for (int t = 0; t < 2; ++t) {
            const int r  = wv*32 + t*16 + ln;
            const int c0 = hl*16;
            #pragma unroll
            for (int j = 0; j < 16; ++j) a[t][j] = lds_x[r][c0+j];
        }
        // ---- 4 B fragments (32x16): lanes<16 K=0..15, lanes>=16 K=16..31;
        //      each feeds both row sub-tiles ----
        #pragma unroll
        for (int nt = 0; nt < 4; ++nt) {
            v16h b;
            const int n  = nt*16 + ln;
            const int c0 = hl*16;
            #pragma unroll
            for (int j = 0; j < 16; ++j) b[j] = lds_w[n][c0+j];
            #pragma unroll
            for (int t = 0; t < 2; ++t)
                acc[t][nt] = __builtin_amdgcn_wmma_f32_16x16x32_f16(
                    false, a[t], false, b, (short)0, acc[t][nt], false, false);
        }
        __syncthreads();
    }

    // ---- store C (f32 accum -> f16 ws); C layout: VGPR r -> row hl*8+r, col=lane ----
    #pragma unroll
    for (int t = 0; t < 2; ++t) {
        #pragma unroll
        for (int nt = 0; nt < 4; ++nt) {
            const int col = colbase + nt*16 + ln;
            #pragma unroll
            for (int r = 0; r < 8; ++r) {
                const int row = rowbase + wv*32 + t*16 + hl*8 + r;
                Out[(size_t)row * EMB + col] = (_Float16)acc[t][nt][r];
            }
        }
    }
}

// ---------------------------------------------------------------------------
// Stage 2: causal softcapped flash attention, fixed-shift softmax.
// S = 30*tanh(QK^T/8) <= 30  =>  softmax shift is the constant 30: no online
// max / rescaling needed.  Block = (batch, head, 64 q rows), 4 waves; wave w
// owns q rows w*16..+15.  K tile staged via the Tensor Data Mover when the
// toolchain exposes it (TENSORcnt path), V staged transposed manually.
// ---------------------------------------------------------------------------
__global__ __launch_bounds__(128)
void flash_attn_kernel(const _Float16* __restrict__ ws_qkv,
                       const int* __restrict__ amask,
                       float* __restrict__ Out)
{
    __shared__ _Float16 lds_k [64][64];      // [key][d]   (B-frag for QK^T: contiguous)
    __shared__ _Float16 lds_vT[64][64];      // [d][key]   (B-frag for P*V:  contiguous)
    __shared__ _Float16 lds_p [4][16][64];   // per-wave P tile [row][key]

    const _Float16* Q = ws_qkv;
    const _Float16* K = ws_qkv + (size_t)1 * BLROWS * EMB;
    const _Float16* V = ws_qkv + (size_t)2 * BLROWS * EMB;

    const int tid  = threadIdx.x;
    const int wv   = tid >> 5;
    const int lane = tid & 31;
    const int hl   = lane >> 4;
    const int ln   = lane & 15;

    const int qt = blockIdx.x;       // 64-row query tile
    const int h  = blockIdx.y;
    const int bz = blockIdx.z;

    const int q0   = qt*64 + wv*16;  // wave's first query row (seq position)
    const int ecol = h * DHEAD;

    // ---- load this wave's Q A-fragments once (Dh=64 -> 2 k-steps of 32) ----
    v16h qa[2];
    {
        const _Float16* qrow = Q + (size_t)(bz*SEQ + q0 + ln) * EMB + ecol;
        #pragma unroll
        for (int s = 0; s < 2; ++s) {
            const int c0 = s*32 + hl*16;
            #pragma unroll
            for (int j = 0; j < 16; ++j) qa[s][j] = qrow[c0+j];
        }
    }

    v8f   o[4] = {};                 // output accumulators, 16x64 per wave
    float lsum[8];
    #pragma unroll
    for (int r = 0; r < 8; ++r) lsum[r] = 0.0f;

    const int nkb  = qt + 1;         // causal: key tiles 0..qt
    const int srow = tid >> 1;       // staging key 0..63
    const int sd   = (tid & 1) * 32; // staging d offset

    for (int kb = 0; kb < nkb; ++kb) {
        // ---- stage K tile [key][d] ----
#ifdef USE_TDM
        if (tid == 0) {
            // Tensor DMA descriptor (D#), 2-D tile: 64 keys x 64 halves.
            const unsigned long long ga = (unsigned long long)
                (K + (size_t)(bz*SEQ + kb*64) * EMB + ecol);
            const unsigned int ldsa = (unsigned int)(unsigned long long)&lds_k[0][0];
            v4u g0 = { 1u,                                   // count=1, user mode
                       ldsa,                                 // lds_addr
                       (unsigned int)ga,                     // global_addr[31:0]
                       (unsigned int)((ga >> 32) & 0x01FFFFFFu) | (2u << 30) }; // addr[56:32] | type=2
            v8i g1 = { (int)0x00010000,        // wg_mask=0, data_size=1 (2 bytes)
                       (int)(1024u << 16),     // tensor_dim0 = 1024 (row length)
                       (int)(4096u << 16),     // tensor_dim1 = 4096 (rows)
                       (int)(64u   << 16),     // tile_dim0 = 64
                       64,                     // tile_dim1 = 64, tile_dim2 = 0
                       1024,                   // tensor_dim0_stride = EMB
                       0, 0 };
            v4i gz = {};
#if defined(__clang_major__) && __clang_major__ >= 23
            v8i gz8 = {};
            __builtin_amdgcn_tensor_load_to_lds(g0, g1, gz, gz, gz8, 0);
#else
            __builtin_amdgcn_tensor_load_to_lds(g0, g1, gz, gz, 0);
#endif
        }
#else
        {
            const _Float16* ksrc = K + (size_t)(bz*SEQ + kb*64 + srow) * EMB + ecol + sd;
            #pragma unroll
            for (int i = 0; i < 32; i += 8)
                *(uint4*)&lds_k[srow][sd+i] = *(const uint4*)(ksrc + i);
        }
#endif
        // ---- stage V transposed [d][key] (TDM cannot transpose) ----
        {
            const _Float16* vsrc = V + (size_t)(bz*SEQ + kb*64 + srow) * EMB + ecol + sd;
            #pragma unroll
            for (int i = 0; i < 32; ++i)
                lds_vT[sd+i][srow] = vsrc[i];
        }
        if (kb + 1 < nkb) {   // warm caches for next V tile
            __builtin_prefetch(V + (size_t)(bz*SEQ + (kb+1)*64 + srow) * EMB + ecol, 0, 0);
        }
#ifdef USE_TDM
        __builtin_amdgcn_s_wait_tensorcnt(0);   // issuing wave drains TDM; others no-op
#endif
        __syncthreads();

        // ---- S = Q K^T over 4 key sub-tiles; softcap + mask + exp(. - 30) ----
        #pragma unroll
        for (int nt = 0; nt < 4; ++nt) {
            v8f s = {};
            #pragma unroll
            for (int ks = 0; ks < 2; ++ks) {
                v16h b;
                const int keyl = nt*16 + ln;
                const int c0   = ks*32 + hl*16;
                #pragma unroll
                for (int j = 0; j < 16; ++j) b[j] = lds_k[keyl][c0+j];
                s = __builtin_amdgcn_wmma_f32_16x16x32_f16(
                        false, qa[ks], false, b, (short)0, s, false, false);
            }
            const int key  = kb*64 + nt*16 + ln;
            const int keep = amask[bz*SEQ + key];
            #pragma unroll
            for (int r = 0; r < 8; ++r) {
                const int qrow = q0 + hl*8 + r;
                float sv = TAUF * tanhf(s[r] * 0.125f);           // softcap, scale 1/sqrt(64)
                float p  = (key <= qrow && keep != 0) ? __expf(sv - TAUF) : 0.0f;
                lsum[r] += p;
                lds_p[wv][hl*8 + r][nt*16 + ln] = (_Float16)p;
            }
        }

        // ---- O += P * V  (P re-swizzled through LDS into A-operand layout) ----
        #pragma unroll
        for (int ks = 0; ks < 2; ++ks) {
            v16h pa;
            const int c0 = ks*32 + hl*16;
            #pragma unroll
            for (int j = 0; j < 16; ++j) pa[j] = lds_p[wv][ln][c0+j];
            #pragma unroll
            for (int nt = 0; nt < 4; ++nt) {
                v16h vb;
                const int d = nt*16 + ln;
                const int c = ks*32 + hl*16;
                #pragma unroll
                for (int j = 0; j < 16; ++j) vb[j] = lds_vT[d][c+j];
                o[nt] = __builtin_amdgcn_wmma_f32_16x16x32_f16(
                            false, pa, false, vb, (short)0, o[nt], false, false);
            }
        }
        __syncthreads();
    }

    // ---- row sums: reduce across the 16 lanes holding each row's columns ----
    #pragma unroll
    for (int r = 0; r < 8; ++r) {
        float v = lsum[r];
        #pragma unroll
        for (int off = 1; off < 16; off <<= 1) v += __shfl_xor(v, off, 32);
        lsum[r] = v;
    }

    // ---- normalize and store f32 output ----
    #pragma unroll
    for (int nt = 0; nt < 4; ++nt) {
        const int d = nt*16 + ln;
        #pragma unroll
        for (int r = 0; r < 8; ++r) {
            const int qrow = q0 + hl*8 + r;
            Out[(size_t)(bz*SEQ + qrow) * EMB + ecol + d] = o[nt][r] / lsum[r];
        }
    }
}

// ---------------------------------------------------------------------------
extern "C" void kernel_launch(void* const* d_in, const int* in_sizes, int n_in,
                              void* d_out, int out_size, void* d_ws, size_t ws_size,
                              hipStream_t stream)
{
    (void)in_sizes; (void)n_in; (void)out_size; (void)ws_size;
    const float* X  = (const float*)d_in[0];
    const int*   am = (const int*)  d_in[1];
    const float* Wq = (const float*)d_in[2];
    const float* Wk = (const float*)d_in[3];
    const float* Wv = (const float*)d_in[4];
    float* out      = (float*)d_out;
    _Float16* ws_qkv = (_Float16*)d_ws;    // Q|K|V f16: 3 * 4096 * 1024 * 2B = 24 MiB

    dim3 g1(EMB/64, BLROWS/128, 3);        // 16 x 32 x 3 tiles
    qkv_gemm_kernel<<<g1, 128, 0, stream>>>(X, Wq, Wk, Wv, ws_qkv);

    dim3 g2(SEQ/64, NHEADS, BATCH);        // 32 x 16 x 2
    flash_attn_kernel<<<g2, 128, 0, stream>>>(ws_qkv, am, out);
}
